// TPatchGNN_22222160789768
// MI455X (gfx1250) — matrix operand order repaired and checked by reading, not compiled, not run on hardware
//
#include <hip/hip_runtime.h>
#include <hip/hip_bf16.h>
#include <math.h>

typedef __attribute__((ext_vector_type(16))) _Float16 v16h;
typedef __attribute__((ext_vector_type(8)))  _Float16 v8h;
typedef __attribute__((ext_vector_type(8)))  float    v8f;

#define B_    128
#define N_    5
#define L_    128
#define S_    8
#define P_    16
#define D_    64
#define DT_   16
#define DIN_  17
#define NF_   63
#define DG_   16
#define NH_   4
#define NC_   39

__device__ __forceinline__ v8f wmma16x16x32(v16h a, v16h b, v8f c) {
  // D = A(16x32 f16) * B(32x16 f16) + C(16x16 f32)
  return __builtin_amdgcn_wmma_f32_16x16x32_f16(false, a, false, b, (short)0, c, false, false);
}

// A fragment: 16x32 f16 tile, row-major, leading dim ld (ld multiple of 8).
// lanes 0-15: row=lane, K 0-7 & 16-23 ; lanes 16-31: row=lane-16, K 8-15 & 24-31
// Per lane: two contiguous 16B vector loads.
__device__ __forceinline__ v16h frag_a(const _Float16* src, int ld) {
  int lane = threadIdx.x & 31;
  int row = lane & 15;
  int kb = (lane & 16) ? 8 : 0;
  const _Float16* p = src + row * ld + kb;
  v8h lo = *(const v8h*)p;
  v8h hi = *(const v8h*)(p + 16);
  v16h a;
#pragma unroll
  for (int i = 0; i < 8; ++i) { a[i] = lo[i]; a[8 + i] = hi[i]; }
  return a;
}

// B fragment from f16 weight stored row-major (O x K): B[k][n] = W[n0+n][k0+k].
// Per lane: K 8h..8h+7 and 16+8h..16+8h+7 (matches frag_a) -> two 16B vector loads.
__device__ __forceinline__ v16h frag_b16(const _Float16* W, int ldk, int n0, int k0) {
  int lane = threadIdx.x & 31;
  int col = lane & 15;
  int kb = (lane & 16) ? 8 : 0;
  const _Float16* p = W + (size_t)(n0 + col) * ldk + k0 + kb;
  v8h lo = *(const v8h*)p;
  v8h hi = *(const v8h*)(p + 16);
  v16h b;
#pragma unroll
  for (int i = 0; i < 8; ++i) { b[i] = lo[i]; b[8 + i] = hi[i]; }
  return b;
}

// ---------------------------------------------------------------------------
// Weight pre-conversion fp32 -> f16 (runs once per launch; weights then live
// in L2 at half the bytes and B-fragments become pure vector loads)
// ---------------------------------------------------------------------------
__global__ void __launch_bounds__(256) cvt_f16_kernel(
    const float* __restrict__ src, _Float16* __restrict__ dst, int n) {
  int i = blockIdx.x * 256 + threadIdx.x;
  if (i < n) dst[i] = (_Float16)src[i];
}

// ---------------------------------------------------------------------------
// Stage 1: time embedding + TTCN + mask channel + positional encoding
// one 64-thread block per token (b,n,p); thread f handles filter f
// ---------------------------------------------------------------------------
__global__ void __launch_bounds__(64) ttcn_kernel(
    const float* __restrict__ x, const float* __restrict__ t_rel,
    const float* __restrict__ omega, const float* __restrict__ alpha,
    const float* __restrict__ w1, const float* __restrict__ b1,
    const float* __restrict__ w2, const float* __restrict__ b2,
    const float* __restrict__ pos, float* __restrict__ h)
{
  __shared__ float Z[S_ * DIN_];           // 8 x 17 shared token features
  __shared__ float LG[NF_ * S_ * DIN_];    // per-filter logits (softmax over s)

  const int tid = threadIdx.x;
  const int tok = blockIdx.x;
  const int b   = tok / (N_ * P_);
  const int rem = tok - b * (N_ * P_);
  const int n   = rem / P_;
  const int p   = rem - n * P_;

  for (int i = tid; i < S_ * DIN_; i += 64) {
    int s = i / DIN_, c = i - s * DIN_;
    float t = t_rel[b * L_ + p * S_ + s];
    float v;
    if (c == 0)       v = omega[0] * t + alpha[0];
    else if (c < DT_) v = __sinf(omega[c] * t + alpha[c]);
    else              v = x[((size_t)b * N_ + n) * L_ + p * S_ + s];
    Z[i] = v;
  }
  __syncthreads();

  const int f = tid;
  if (f < NF_) {
    for (int s = 0; s < S_; ++s) {
      float hj[DIN_];
      for (int j = 0; j < DIN_; ++j) {
        float a = b1[f * DIN_ + j];
        const float* wr = w1 + ((size_t)f * DIN_ + j) * DIN_;
        for (int i = 0; i < DIN_; ++i) a += Z[s * DIN_ + i] * wr[i];
        hj[j] = fmaxf(a, 0.0f);
      }
      for (int k = 0; k < DIN_; ++k) {
        float a = b2[f * DIN_ + k];
        const float* wr = w2 + ((size_t)f * DIN_ + k) * DIN_;
        for (int j = 0; j < DIN_; ++j) a += hj[j] * wr[j];
        LG[(f * S_ + s) * DIN_ + k] = a;
      }
    }
    float out = 0.0f;
    for (int k = 0; k < DIN_; ++k) {
      float mx = -1e30f;
      for (int s = 0; s < S_; ++s) mx = fmaxf(mx, LG[(f * S_ + s) * DIN_ + k]);
      float sum = 0.0f;
      for (int s = 0; s < S_; ++s) sum += __expf(LG[(f * S_ + s) * DIN_ + k] - mx);
      float inv = 1.0f / sum;
      for (int s = 0; s < S_; ++s)
        out += __expf(LG[(f * S_ + s) * DIN_ + k] - mx) * inv * Z[s * DIN_ + k];
    }
    h[(((size_t)b * N_ + n) * P_ + p) * D_ + f] = out + pos[p * D_ + f];
  } else if (f == NF_) {
    // mask channel (mask all-true) + pos enc
    h[(((size_t)b * N_ + n) * P_ + p) * D_ + NF_] = 1.0f + pos[p * D_ + NF_];
  }
}

// ---------------------------------------------------------------------------
// Stage 2a: intra-series transformer block. One wave per (b,n) sequence.
// All GEMMs via v_wmma_f32_16x16x32_f16; fp32 residual/LN state in LDS.
// A-fragments hoisted out of output-tile loops; V kept transposed for A*V.
// ---------------------------------------------------------------------------
__global__ void __launch_bounds__(32) xformer_kernel(
    float* __restrict__ h,
    const _Float16* __restrict__ wi, const float* __restrict__ bi,
    const _Float16* __restrict__ wo, const float* __restrict__ bo,
    const float* __restrict__ g1, const float* __restrict__ be1,
    const _Float16* __restrict__ fw1, const float* __restrict__ fb1,
    const _Float16* __restrict__ fw2, const float* __restrict__ fb2,
    const float* __restrict__ g2, const float* __restrict__ be2)
{
  __shared__ float                Xf[P_ * D_];      // fp32 residual state
  __shared__ alignas(16) _Float16 Xh[P_ * D_];      // f16 mirror (WMMA A)
  __shared__ alignas(16) _Float16 QK[P_ * 2 * D_];  // q|k rows (row-major)
  __shared__ alignas(16) _Float16 Vt[D_ * P_];      // V transposed [feat][key]
  __shared__ alignas(16) _Float16 Aw[P_ * P_];      // attn weights (one head)
  __shared__ alignas(16) _Float16 Oh[P_ * D_];      // attn context
  __shared__ float                Tf[P_ * D_];      // fp32 scratch
  __shared__ alignas(16) _Float16 H1[P_ * 2 * D_];  // ffn hidden

  const int lane = threadIdx.x;
  float* hrow = h + (size_t)blockIdx.x * P_ * D_;

  for (int i = lane; i < P_ * D_; i += 32) {
    float v = hrow[i];
    Xf[i] = v; Xh[i] = (_Float16)v;
  }
  __syncthreads();

  const int col = lane & 15;
  const int rb  = (lane & 16) ? 8 : 0;

  // ---- QKV projection: (16x64) x (64x192); A-frags loop-invariant ----
  {
    v16h a0 = frag_a(Xh, D_);
    v16h a1 = frag_a(Xh + 32, D_);
    for (int mt = 0; mt < 12; ++mt) {
      v8f acc = {};
      acc = wmma16x16x32(a0, frag_b16(wi, D_, mt * 16, 0), acc);
      acc = wmma16x16x32(a1, frag_b16(wi, D_, mt * 16, 32), acc);
      int o = mt * 16 + col;
      float bias = bi[o];
      if (mt < 8) {                       // Q,K: row-major
#pragma unroll
        for (int v = 0; v < 8; ++v)
          QK[(rb + v) * 128 + o] = (_Float16)(acc[v] + bias);
      } else {                            // V: transposed (feature-major)
        v8h pk;
#pragma unroll
        for (int v = 0; v < 8; ++v) pk[v] = (_Float16)(acc[v] + bias);
        *(v8h*)(Vt + (o - 128) * P_ + rb) = pk;   // 16B-aligned vector store
      }
    }
  }
  __syncthreads();

  // ---- attention per head (dh=16, K padded 16->32 with zeros) ----
  for (int hd = 0; hd < NH_; ++hd) {
    const int hb = hd * 16;
    // scores = Q Kt
    v16h aq, bkf;
    {
      int row = lane & 15;
      int kq  = (lane & 16) ? 8 : 0;
      v8h q = *(const v8h*)(QK + row * 128 + hb + kq);
#pragma unroll
      for (int i = 0; i < 8; ++i) { aq[i] = q[i]; aq[8 + i] = (_Float16)0.0f; }
      {
        v8h k0 = *(const v8h*)(QK + col * 128 + D_ + hb + kq);
#pragma unroll
        for (int i = 0; i < 8; ++i) { bkf[i] = k0[i]; bkf[8 + i] = (_Float16)0.0f; }
      }
    }
    v8f sc = {};
    sc = wmma16x16x32(aq, bkf, sc);

    // row softmax: each query row lives in one 16-lane group -> shfl_xor(16)
#pragma unroll
    for (int v = 0; v < 8; ++v) {
      float s = sc[v] * 0.25f;               // 1/sqrt(16)
      float m = s;
#pragma unroll
      for (int o = 8; o; o >>= 1) m = fmaxf(m, __shfl_xor(m, o, 16));
      float e = __expf(s - m);
      float sum = e;
#pragma unroll
      for (int o = 8; o; o >>= 1) sum += __shfl_xor(sum, o, 16);
      Aw[(rb + v) * 16 + col] = (_Float16)(e / sum);
    }
    __syncthreads();

    // context = Aw (16x16) @ V_head; B from transposed V -> vector loads
    v16h aa, bv;
    {
      int row = lane & 15;
      int kq  = (lane & 16) ? 8 : 0;
      v8h a = *(const v8h*)(Aw + row * 16 + kq);
#pragma unroll
      for (int i = 0; i < 8; ++i) { aa[i] = a[i]; aa[8 + i] = (_Float16)0.0f; }
      {
        v8h v0 = *(const v8h*)(Vt + (hb + col) * P_ + kq);
#pragma unroll
        for (int i = 0; i < 8; ++i) { bv[i] = v0[i]; bv[8 + i] = (_Float16)0.0f; }
      }
    }
    v8f oc = {};
    oc = wmma16x16x32(aa, bv, oc);
#pragma unroll
    for (int v = 0; v < 8; ++v)
      Oh[(rb + v) * D_ + hb + col] = (_Float16)oc[v];
    __syncthreads();
  }

  // ---- output projection ----
  {
    v16h a0 = frag_a(Oh, D_);
    v16h a1 = frag_a(Oh + 32, D_);
    for (int mt = 0; mt < 4; ++mt) {
      v8f acc = {};
      acc = wmma16x16x32(a0, frag_b16(wo, D_, mt * 16, 0), acc);
      acc = wmma16x16x32(a1, frag_b16(wo, D_, mt * 16, 32), acc);
      int o = mt * 16 + col;
      float bias = bo[o];
#pragma unroll
      for (int v = 0; v < 8; ++v)
        Tf[(rb + v) * D_ + o] = acc[v] + bias;
    }
  }
  __syncthreads();

  // ---- residual + LN1 (one row per lane, lanes 0..15) ----
  if (lane < 16) {
    float mu = 0.f;
    for (int d = 0; d < D_; ++d) { float y = Xf[lane*D_+d] + Tf[lane*D_+d]; Tf[lane*D_+d] = y; mu += y; }
    mu *= (1.0f / D_);
    float var = 0.f;
    for (int d = 0; d < D_; ++d) { float dd = Tf[lane*D_+d] - mu; var += dd*dd; }
    var *= (1.0f / D_);
    float rs = rsqrtf(var + 1e-5f);
    for (int d = 0; d < D_; ++d) {
      float y = (Tf[lane*D_+d] - mu) * rs * g1[d] + be1[d];
      Xf[lane*D_+d] = y; Xh[lane*D_+d] = (_Float16)y;
    }
  }
  __syncthreads();

  // ---- FFN: (16x64)->(16x128)->(16x64) ----
  {
    v16h a0 = frag_a(Xh, D_);
    v16h a1 = frag_a(Xh + 32, D_);
    for (int mt = 0; mt < 8; ++mt) {
      v8f acc = {};
      acc = wmma16x16x32(a0, frag_b16(fw1, D_, mt * 16, 0), acc);
      acc = wmma16x16x32(a1, frag_b16(fw1, D_, mt * 16, 32), acc);
      int o = mt * 16 + col;
      float bias = fb1[o];
#pragma unroll
      for (int v = 0; v < 8; ++v)
        H1[(rb + v) * 128 + o] = (_Float16)fmaxf(acc[v] + bias, 0.0f);
    }
  }
  __syncthreads();
  {
    v16h f0 = frag_a(H1, 128);
    v16h f1 = frag_a(H1 + 32, 128);
    v16h f2 = frag_a(H1 + 64, 128);
    v16h f3 = frag_a(H1 + 96, 128);
    for (int mt = 0; mt < 4; ++mt) {
      v8f acc = {};
      acc = wmma16x16x32(f0, frag_b16(fw2, 128, mt * 16, 0), acc);
      acc = wmma16x16x32(f1, frag_b16(fw2, 128, mt * 16, 32), acc);
      acc = wmma16x16x32(f2, frag_b16(fw2, 128, mt * 16, 64), acc);
      acc = wmma16x16x32(f3, frag_b16(fw2, 128, mt * 16, 96), acc);
      int o = mt * 16 + col;
      float bias = fb2[o];
#pragma unroll
      for (int v = 0; v < 8; ++v)
        Tf[(rb + v) * D_ + o] = acc[v] + bias;
    }
  }
  __syncthreads();

  // ---- residual + LN2, write back ----
  if (lane < 16) {
    float mu = 0.f;
    for (int d = 0; d < D_; ++d) { float y = Xf[lane*D_+d] + Tf[lane*D_+d]; Tf[lane*D_+d] = y; mu += y; }
    mu *= (1.0f / D_);
    float var = 0.f;
    for (int d = 0; d < D_; ++d) { float dd = Tf[lane*D_+d] - mu; var += dd*dd; }
    var *= (1.0f / D_);
    float rs = rsqrtf(var + 1e-5f);
    for (int d = 0; d < D_; ++d)
      hrow[lane*D_+d] = (Tf[lane*D_+d] - mu) * rs * g2[d] + be2[d];
  }
}

// ---------------------------------------------------------------------------
// Stage 2b: inter-series GSL + GNN. One wave per (b,p) graph of N_=5 nodes.
// 5-node shapes don't amortize 16x16 WMMA tiles -> VALU.
// ---------------------------------------------------------------------------
__global__ void __launch_bounds__(32) gsl_kernel(
    float* __restrict__ h,
    const float* __restrict__ Es1, const float* __restrict__ Es2,
    const float* __restrict__ Wd1, const float* __restrict__ bd1,
    const float* __restrict__ Wd2, const float* __restrict__ bd2,
    const float* __restrict__ Wg1, const float* __restrict__ bg1,
    const float* __restrict__ Wg2, const float* __restrict__ bg2,
    const float* __restrict__ gW, const float* __restrict__ gb,
    const float* __restrict__ g3, const float* __restrict__ be3)
{
  __shared__ float HH[N_ * D_];
  __shared__ float ED1[N_ * DG_], ED2[N_ * DG_];
  __shared__ float EP1[N_ * DG_], EP2[N_ * DG_];
  __shared__ float GG[2 * N_];
  __shared__ float Amat[N_ * N_], A2[N_ * N_];
  __shared__ float T1[N_ * D_], T2[N_ * D_];
  __shared__ float RES[N_ * D_];
  __shared__ float MUV[2 * N_];

  const int lane = threadIdx.x;
  const int bp = blockIdx.x;
  const int b = bp >> 4;
  const int p = bp & 15;

  for (int i = lane; i < N_ * D_; i += 32) {
    int n = i >> 6, d = i & 63;
    HH[i] = h[(((size_t)b * N_ + n) * P_ + p) * D_ + d];
  }
  __syncthreads();

  for (int i = lane; i < N_ * DG_; i += 32) {
    int n = i >> 4, g = i & 15;
    float a1 = bd1[g], a2 = bd2[g];
    for (int d = 0; d < D_; ++d) {
      float xv = HH[n * D_ + d];
      a1 += xv * Wd1[g * D_ + d];
      a2 += xv * Wd2[g * D_ + d];
    }
    ED1[i] = a1; ED2[i] = a2;
  }
  if (lane < 2 * N_) {
    int which = lane / N_, n = lane % N_;
    const float* Wg = which ? Wg2 : Wg1;
    const float* Es = which ? Es2 : Es1;
    float a = which ? bg2[0] : bg1[0];
    for (int d = 0; d < D_; ++d) a += HH[n * D_ + d] * Wg[d];
    for (int g = 0; g < DG_; ++g) a += Es[n * DG_ + g] * Wg[D_ + g];
    GG[lane] = fmaxf(tanhf(a), 0.0f);
  }
  __syncthreads();
  for (int i = lane; i < N_ * DG_; i += 32) {
    int n = i >> 4;
    EP1[i] = Es1[i] + GG[n] * ED1[i];
    EP2[i] = Es2[i] + GG[N_ + n] * ED2[i];
  }
  __syncthreads();
  if (lane < N_ * N_) {
    int n = lane / N_, m = lane % N_;
    float a = 0.f;
    for (int g = 0; g < DG_; ++g) a += EP1[n * DG_ + g] * EP2[m * DG_ + g];
    Amat[lane] = fmaxf(a, 0.0f);
  }
  __syncthreads();
  if (lane < N_) {
    float mx = -1e30f;
    for (int m = 0; m < N_; ++m) mx = fmaxf(mx, Amat[lane * N_ + m]);
    float s = 0.f;
    for (int m = 0; m < N_; ++m) { float e = __expf(Amat[lane * N_ + m] - mx); Amat[lane * N_ + m] = e; s += e; }
    float inv = 1.0f / s;
    for (int m = 0; m < N_; ++m) Amat[lane * N_ + m] *= inv;
  }
  __syncthreads();
  for (int i = lane; i < N_ * D_; i += 32) {
    int n = i >> 6, d = i & 63;
    float a = 0.f;
    for (int m = 0; m < N_; ++m) a += Amat[n * N_ + m] * HH[m * D_ + d];
    T1[i] = a;
  }
  if (lane < N_ * N_) {
    int n = lane / N_, m = lane % N_;
    float a = 0.f;
    for (int j = 0; j < N_; ++j) a += Amat[n * N_ + j] * Amat[j * N_ + m];
    A2[lane] = a;
  }
  __syncthreads();
  for (int i = lane; i < N_ * D_; i += 32) {
    int n = i >> 6, d = i & 63;
    float a = 0.f;
    for (int m = 0; m < N_; ++m) a += A2[n * N_ + m] * HH[m * D_ + d];
    T2[i] = a;
  }
  __syncthreads();
  for (int i = lane; i < N_ * D_; i += 32) {
    int n = i >> 6, d = i & 63;
    float a = gb[d] + gb[D_ + d] + gb[2 * D_ + d];
    const float* w0 = gW + (size_t)d * D_;
    const float* w1 = gW + (size_t)D_ * D_ + (size_t)d * D_;
    const float* w2 = gW + (size_t)2 * D_ * D_ + (size_t)d * D_;
    for (int k = 0; k < D_; ++k) {
      a += HH[n * D_ + k] * w0[k];
      a += T1[n * D_ + k] * w1[k];
      a += T2[n * D_ + k] * w2[k];
    }
    RES[i] = HH[i] + fmaxf(a, 0.0f);
  }
  __syncthreads();
  if (lane < N_) {
    float mu = 0.f;
    for (int d = 0; d < D_; ++d) mu += RES[lane * D_ + d];
    mu *= (1.0f / D_);
    float var = 0.f;
    for (int d = 0; d < D_; ++d) { float dd = RES[lane * D_ + d] - mu; var += dd * dd; }
    var *= (1.0f / D_);
    MUV[lane] = mu; MUV[N_ + lane] = rsqrtf(var + 1e-5f);
  }
  __syncthreads();
  for (int i = lane; i < N_ * D_; i += 32) {
    int n = i >> 6, d = i & 63;
    float y = (RES[i] - MUV[n]) * MUV[N_ + n] * g3[d] + be3[d];
    h[(((size_t)b * N_ + n) * P_ + p) * D_ + d] = y;
  }
}

// ---------------------------------------------------------------------------
// Stage 3: flatten projection (640x1024 @ 1024x64 via WMMA, f16 weights) +
// query embed + per-variable 3-layer head. One wave per 16-row tile.
// ks-major loop: each A-fragment loaded once, 4 live accumulators.
// ---------------------------------------------------------------------------
__global__ void __launch_bounds__(32) head_kernel(
    const float* __restrict__ h,
    const _Float16* __restrict__ fp_w16, const float* __restrict__ fp_b,
    const float* __restrict__ qo, const float* __restrict__ qa,
    const float* __restrict__ ow1, const float* __restrict__ ob1,
    const float* __restrict__ ow2, const float* __restrict__ ob2,
    const float* __restrict__ ow3, const float* __restrict__ ob3,
    float* __restrict__ out)
{
  __shared__ alignas(16) _Float16 Af[16 * 1024];   // 16 flat rows, f16 staging
  __shared__ float Hs[16 * D_];

  const int lane = threadIdx.x;
  const int r0 = blockIdx.x * 16;
  const float* src = h + (size_t)r0 * (P_ * D_);
  for (int i = lane; i < 16 * 1024; i += 32) Af[i] = (_Float16)src[i];
  __syncthreads();

  const int col = lane & 15;
  const int rb  = (lane & 16) ? 8 : 0;
  v8f acc0 = {}, acc1 = {}, acc2 = {}, acc3 = {};
  for (int ks = 0; ks < 32; ++ks) {
    v16h a = frag_a(Af + ks * 32, 1024);
    acc0 = wmma16x16x32(a, frag_b16(fp_w16, 1024,  0, ks * 32), acc0);
    acc1 = wmma16x16x32(a, frag_b16(fp_w16, 1024, 16, ks * 32), acc1);
    acc2 = wmma16x16x32(a, frag_b16(fp_w16, 1024, 32, ks * 32), acc2);
    acc3 = wmma16x16x32(a, frag_b16(fp_w16, 1024, 48, ks * 32), acc3);
  }
#pragma unroll
  for (int v = 0; v < 8; ++v) {
    Hs[(rb + v) * D_ +  0 + col] = acc0[v] + fp_b[ 0 + col];
    Hs[(rb + v) * D_ + 16 + col] = acc1[v] + fp_b[16 + col];
    Hs[(rb + v) * D_ + 32 + col] = acc2[v] + fp_b[32 + col];
    Hs[(rb + v) * D_ + 48 + col] = acc3[v] + fp_b[48 + col];
  }
  __syncthreads();

  if (lane < 16) {
    int r = r0 + lane;
    int n = r % N_;
    float inp[D_ + DT_];
    for (int d = 0; d < D_; ++d) inp[d] = Hs[lane * D_ + d];
    const float t = (float)L_;
    inp[D_] = qo[0] * t + qa[0];
    for (int c = 1; c < DT_; ++c) inp[D_ + c] = __sinf(qo[c] * t + qa[c]);

    float h1[D_];
    const float* W1 = ow1 + (size_t)n * D_ * (D_ + DT_);
    for (int o = 0; o < D_; ++o) {
      float a = ob1[n * D_ + o];
      const float* wr = W1 + (size_t)o * (D_ + DT_);
      for (int i = 0; i < D_ + DT_; ++i) a += inp[i] * wr[i];
      h1[o] = fmaxf(a, 0.0f);
    }
    float h2[D_];
    const float* W2 = ow2 + (size_t)n * D_ * D_;
    for (int o = 0; o < D_; ++o) {
      float a = ob2[n * D_ + o];
      const float* wr = W2 + (size_t)o * D_;
      for (int i = 0; i < D_; ++i) a += h1[i] * wr[i];
      h2[o] = fmaxf(a, 0.0f);
    }
    const float* W3 = ow3 + (size_t)n * NC_ * D_;
    for (int o = 0; o < NC_; ++o) {
      float a = ob3[n * NC_ + o];
      const float* wr = W3 + (size_t)o * D_;
      for (int i = 0; i < D_; ++i) a += h2[i] * wr[i];
      out[(size_t)r * NC_ + o] = a;
    }
  }
}

// ---------------------------------------------------------------------------
extern "C" void kernel_launch(void* const* d_in, const int* in_sizes, int n_in,
                              void* d_out, int out_size, void* d_ws, size_t ws_size,
                              hipStream_t stream) {
  const float* x          = (const float*)d_in[0];
  const float* t_rel      = (const float*)d_in[1];
  const float* te_omega   = (const float*)d_in[2];
  const float* te_alpha   = (const float*)d_in[3];
  const float* mf_w1      = (const float*)d_in[4];
  const float* mf_b1      = (const float*)d_in[5];
  const float* mf_w2      = (const float*)d_in[6];
  const float* mf_b2      = (const float*)d_in[7];
  const float* pos_enc    = (const float*)d_in[8];
  const float* attn_in_w  = (const float*)d_in[9];
  const float* attn_in_b  = (const float*)d_in[10];
  const float* attn_out_w = (const float*)d_in[11];
  const float* attn_out_b = (const float*)d_in[12];
  const float* ln1_g      = (const float*)d_in[13];
  const float* ln1_b      = (const float*)d_in[14];
  const float* ffn_w1     = (const float*)d_in[15];
  const float* ffn_b1     = (const float*)d_in[16];
  const float* ffn_w2     = (const float*)d_in[17];
  const float* ffn_b2     = (const float*)d_in[18];
  const float* ln2_g      = (const float*)d_in[19];
  const float* ln2_b      = (const float*)d_in[20];
  const float* Es1        = (const float*)d_in[21];
  const float* Es2        = (const float*)d_in[22];
  const float* Wd1        = (const float*)d_in[23];
  const float* bd1        = (const float*)d_in[24];
  const float* Wd2        = (const float*)d_in[25];
  const float* bd2        = (const float*)d_in[26];
  const float* Wg1        = (const float*)d_in[27];
  const float* bg1        = (const float*)d_in[28];
  const float* Wg2        = (const float*)d_in[29];
  const float* bg2        = (const float*)d_in[30];
  const float* gnn_W      = (const float*)d_in[31];
  const float* gnn_b      = (const float*)d_in[32];
  const float* ln3_g      = (const float*)d_in[33];
  const float* ln3_b      = (const float*)d_in[34];
  const float* fp_w       = (const float*)d_in[35];
  const float* fp_b       = (const float*)d_in[36];
  const float* q_omega    = (const float*)d_in[37];
  const float* q_alpha    = (const float*)d_in[38];
  const float* ow1        = (const float*)d_in[39];
  const float* ob1        = (const float*)d_in[40];
  const float* ow2        = (const float*)d_in[41];
  const float* ob2        = (const float*)d_in[42];
  const float* ow3        = (const float*)d_in[43];
  const float* ob3        = (const float*)d_in[44];
  (void)in_sizes; (void)n_in; (void)out_size; (void)ws_size;

  // workspace layout:
  //   [0, 2.62MB)  : h (B,N,P,D) fp32 activation state
  //   [2.62MB, ..) : f16 copies of GEMM weights (B-operands)
  float* h = (float*)d_ws;
  const size_t hBytes = (size_t)B_ * N_ * P_ * D_ * sizeof(float);
  _Float16* w16 = (_Float16*)((char*)d_ws + hBytes);
  _Float16* wi16  = w16;              // 2*192*64 = 24576
  _Float16* wo16  = w16 + 24576;      // 2*64*64  =  8192
  _Float16* fw116 = w16 + 32768;      // 2*128*64 = 16384
  _Float16* fw216 = w16 + 49152;      // 2*64*128 = 16384
  _Float16* fpw16 = w16 + 65536;      // 64*1024  = 65536

  cvt_f16_kernel<<<(24576 + 255) / 256, 256, 0, stream>>>(attn_in_w,  wi16,  24576);
  cvt_f16_kernel<<<( 8192 + 255) / 256, 256, 0, stream>>>(attn_out_w, wo16,   8192);
  cvt_f16_kernel<<<(16384 + 255) / 256, 256, 0, stream>>>(ffn_w1,     fw116, 16384);
  cvt_f16_kernel<<<(16384 + 255) / 256, 256, 0, stream>>>(ffn_w2,     fw216, 16384);
  cvt_f16_kernel<<<(65536 + 255) / 256, 256, 0, stream>>>(fp_w,       fpw16, 65536);

  ttcn_kernel<<<B_ * N_ * P_, 64, 0, stream>>>(
      x, t_rel, te_omega, te_alpha, mf_w1, mf_b1, mf_w2, mf_b2, pos_enc, h);

  for (int bk = 0; bk < 2; ++bk) {
    xformer_kernel<<<B_ * N_, 32, 0, stream>>>(h,
        wi16  + (size_t)bk * 192 * 64, attn_in_b  + (size_t)bk * 192,
        wo16  + (size_t)bk * 64 * 64,  attn_out_b + (size_t)bk * 64,
        ln1_g + bk * 64, ln1_b + bk * 64,
        fw116 + (size_t)bk * 128 * 64, ffn_b1 + (size_t)bk * 128,
        fw216 + (size_t)bk * 64 * 128, ffn_b2 + (size_t)bk * 64,
        ln2_g + bk * 64, ln2_b + bk * 64);
    gsl_kernel<<<B_ * P_, 32, 0, stream>>>(h,
        Es1 + (size_t)bk * N_ * DG_, Es2 + (size_t)bk * N_ * DG_,
        Wd1 + (size_t)bk * DG_ * D_, bd1 + (size_t)bk * DG_,
        Wd2 + (size_t)bk * DG_ * D_, bd2 + (size_t)bk * DG_,
        Wg1 + (size_t)bk * (D_ + DG_), bg1 + bk,
        Wg2 + (size_t)bk * (D_ + DG_), bg2 + bk,
        gnn_W + (size_t)bk * 3 * D_ * D_, gnn_b + (size_t)bk * 3 * D_,
        ln3_g + bk * 64, ln3_b + bk * 64);
  }

  head_kernel<<<(B_ * N_) / 16, 32, 0, stream>>>(
      h, fpw16, fp_b, q_omega, q_alpha, ow1, ob1, ow2, ob2, ow3, ob3,
      (float*)d_out);
}
